// ISTA_Net_60043642798862
// MI455X (gfx1250) — matrix operand order repaired and checked by reading, not compiled, not run on hardware
//
#include <hip/hip_runtime.h>
#include <hip/hip_bf16.h>

// ---------------------------------------------------------------------------
// ISTA-Net unrolled pipeline for MI455X (gfx1250, wave32, WMMA bf16->f32).
//
// Activations batch-major (2, B, dim) row-major; weights natural row-major:
// every WMMA A-fragment (16 batch rows x 32 K) and B-fragment (output column
// p <- weight row p) is a pair of contiguous 16B loads. K-loops are fully
// unrolled; each kernel keeps exactly two per-lane base pointers and folds
// channel / subtile / k offsets (all < 8MB) into the 24-bit VGLOBAL immediate.
// Per-subtile WMMA issue keeps only two B fragments live -> no spills.
// ---------------------------------------------------------------------------

typedef __attribute__((ext_vector_type(16))) __bf16          v16bf;
typedef __attribute__((ext_vector_type(8)))  float           v8f;
typedef __attribute__((ext_vector_type(8)))  unsigned short  v8u16;
typedef __attribute__((ext_vector_type(8)))  unsigned int    v8u32;
typedef unsigned short u16;

constexpr int Mdim = 1024;   // sparse code size
constexpr int Ndim = 128;    // measurement size
constexpr int Kdim = 256;    // signal size
constexpr int Bdim = 8192;   // batch
constexpr int Titer = 10;

union Frag { v16bf bf; v8u16 h[2]; v8u32 u; };

__device__ __forceinline__ u16 f2bf(float f) {     // native f32 -> bf16 (RNE)
    __bf16 b = (__bf16)f;
    return __builtin_bit_cast(u16, b);
}

// A fragment: pa = per-lane row base (+half*8); off = compile-time element offset.
__device__ __forceinline__ void load_a_off(Frag& f, const u16* pa, int off) {
    f.h[0] = *(const v8u16*)(pa + off);
    f.h[1] = *(const v8u16*)(pa + off + 16);
}
// B fragment: pb = per-lane weight-row base (+half*8).
__device__ __forceinline__ void load_b_off(Frag& f, const u16* pb, int off) {
    f.h[0] = *(const v8u16*)(pb + off);
    f.h[1] = *(const v8u16*)(pb + off + 16);
}

__device__ __forceinline__ v8f wmma(v8f c, const Frag& a, const Frag& b) {
    return __builtin_amdgcn_wmma_f32_16x16x32_bf16(false, a.bf, false, b.bf,
                                                   (short)0, c, false, false);
}

__device__ __forceinline__ float blockSum(float v, float* sh) {
    int t = threadIdx.x;
    sh[t] = v;
    __syncthreads();
    for (int s = 128; s > 0; s >>= 1) {
        if (t < s) sh[t] += sh[t + s];
        __syncthreads();
    }
    float r = sh[0];
    __syncthreads();
    return r;
}

// ---------------------------------------------------------------------------
// GEMM1: T(2,B,N) = cplx(X(2,B,K) * W^T) - y^T ; store bf16.
// ---------------------------------------------------------------------------
__global__ void k_gemm1(const u16* __restrict__ xbf, const u16* __restrict__ wbf,
                        const float* __restrict__ y, u16* __restrict__ tbf) {
    const int tid = blockIdx.x * blockDim.x + threadIdx.x;
    const int wave = tid >> 5, lane = tid & 31;
    const int m = lane & 15, h8 = (lane >> 4) << 3, h16 = (lane >> 4) << 4;
    constexpr int NCOLG = Ndim / 64;
    const int colg = wave % NCOLG, row0 = (wave / NCOLG) * 16;

    constexpr int ACH = Bdim * Kdim;                  // A channel stride (elems)
    constexpr int BCH = Ndim * Kdim;                  // W channel stride
    const u16* pa = xbf + (size_t)(row0 + m) * Kdim + h8;
    const u16* pb = wbf + (size_t)(colg * 64 + m) * Kdim + h8;

    const v8f vz = {0.f,0.f,0.f,0.f,0.f,0.f,0.f,0.f};
    v8f accr[4], acci[4];
#pragma unroll
    for (int s = 0; s < 4; ++s) { accr[s] = vz; acci[s] = vz; }

#pragma unroll
    for (int k0 = 0; k0 < Kdim; k0 += 32) {
        Frag ar, ai, an;
        load_a_off(ar, pa, k0);
        load_a_off(ai, pa, ACH + k0);
        an.u = ai.u ^ 0x80008000u;                    // -Xim
#pragma unroll
        for (int s = 0; s < 4; ++s) {
            Frag b0, b1;
            load_b_off(b0, pb, s * 16 * Kdim + k0);
            load_b_off(b1, pb, BCH + s * 16 * Kdim + k0);
            accr[s] = wmma(accr[s], ar, b0);
            acci[s] = wmma(acci[s], ar, b1);
            accr[s] = wmma(accr[s], an, b1);
            acci[s] = wmma(acci[s], ai, b0);
        }
    }
#pragma unroll
    for (int s = 0; s < 4; ++s) {
        const int c = colg * 64 + s * 16 + m;
#pragma unroll
        for (int j = 0; j < 8; ++j) {
            const int r = row0 + h8 + j;
            const float tr = accr[s][j] - y[((size_t)r * 2 + 0) * Ndim + c];
            const float ti = acci[s][j] - y[((size_t)r * 2 + 1) * Ndim + c];
            tbf[(size_t)r * Ndim + c] = f2bf(tr);
            tbf[(size_t)Bdim * Ndim + (size_t)r * Ndim + c] = f2bf(ti);
        }
    }
}

// ---------------------------------------------------------------------------
// GEMM2: U(2,B,K) = cplx(T * WX^T) ; x <- x - gamma*U ; store f32 + bf16.
// ---------------------------------------------------------------------------
__global__ void k_gemm2(const u16* __restrict__ tbf, const u16* __restrict__ wxbf,
                        const float* __restrict__ gammas, int it,
                        float* __restrict__ x, u16* __restrict__ xbf) {
    const int tid = blockIdx.x * blockDim.x + threadIdx.x;
    const int wave = tid >> 5, lane = tid & 31;
    const int m = lane & 15, h8 = (lane >> 4) << 3, h16 = (lane >> 4) << 4;
    constexpr int NCOLG = Kdim / 64;
    const int colg = wave % NCOLG, row0 = (wave / NCOLG) * 16;

    constexpr int ACH = Bdim * Ndim;                  // T channel stride
    constexpr int BCH = Kdim * Ndim;                  // WX channel stride
    const u16* pa = tbf + (size_t)(row0 + m) * Ndim + h8;
    const u16* pb = wxbf + (size_t)(colg * 64 + m) * Ndim + h8;

    const v8f vz = {0.f,0.f,0.f,0.f,0.f,0.f,0.f,0.f};
    v8f accr[4], acci[4];
#pragma unroll
    for (int s = 0; s < 4; ++s) { accr[s] = vz; acci[s] = vz; }

#pragma unroll
    for (int k0 = 0; k0 < Ndim; k0 += 32) {
        Frag ar, ai, an;
        load_a_off(ar, pa, k0);
        load_a_off(ai, pa, ACH + k0);
        an.u = ai.u ^ 0x80008000u;
#pragma unroll
        for (int s = 0; s < 4; ++s) {
            Frag b0, b1;
            load_b_off(b0, pb, s * 16 * Ndim + k0);
            load_b_off(b1, pb, BCH + s * 16 * Ndim + k0);
            accr[s] = wmma(accr[s], ar, b0);
            acci[s] = wmma(acci[s], ar, b1);
            accr[s] = wmma(accr[s], an, b1);
            acci[s] = wmma(acci[s], ai, b0);
        }
    }
    const float g = gammas[it];
#pragma unroll
    for (int s = 0; s < 4; ++s) {
        const int c = colg * 64 + s * 16 + m;
#pragma unroll
        for (int j = 0; j < 8; ++j) {
            const int r = row0 + h8 + j;
            const size_t ir = (size_t)r * Kdim + c;
            const size_t ii = (size_t)Bdim * Kdim + ir;
            const float xr = x[ir] - g * accr[s][j];
            const float xi = x[ii] - g * acci[s][j];
            x[ir] = xr;  xbf[ir] = f2bf(xr);
            x[ii] = xi;  xbf[ii] = f2bf(xi);
        }
    }
}

// ---------------------------------------------------------------------------
// GEMM3: xsp_re(B,M) = Xre*E0[0]^T - Xim*E0[1]^T ; softshrink(eta) ; losses.
// (Imaginary shrunk channel is never consumed by the reference -> skipped.)
// ---------------------------------------------------------------------------
__global__ void k_gemm3(const u16* __restrict__ xbf, const u16* __restrict__ e0bf,
                        const float* __restrict__ etas, int it,
                        u16* __restrict__ s0bf, float* __restrict__ loss_sparse,
                        float* __restrict__ out_sparse, int last) {
    __shared__ float sh[256];
    const int tid = blockIdx.x * blockDim.x + threadIdx.x;
    const int wave = tid >> 5, lane = tid & 31;
    const int m = lane & 15, h8 = (lane >> 4) << 3, h16 = (lane >> 4) << 4;
    constexpr int NCOLG = Mdim / 64;
    const int colg = wave % NCOLG, row0 = (wave / NCOLG) * 16;

    constexpr int ACH = Bdim * Kdim;                  // X channel stride
    constexpr int BCH = Mdim * Kdim;                  // E0 channel stride
    const u16* pa = xbf + (size_t)(row0 + m) * Kdim + h8;
    const u16* pb = e0bf + (size_t)(colg * 64 + m) * Kdim + h8;

    const v8f vz = {0.f,0.f,0.f,0.f,0.f,0.f,0.f,0.f};
    v8f acc[4];
#pragma unroll
    for (int s = 0; s < 4; ++s) acc[s] = vz;

#pragma unroll
    for (int k0 = 0; k0 < Kdim; k0 += 32) {
        Frag ar, an;
        load_a_off(ar, pa, k0);
        load_a_off(an, pa, ACH + k0);
        an.u = an.u ^ 0x80008000u;                    // -Xim
#pragma unroll
        for (int s = 0; s < 4; ++s) {
            Frag b0, b1;
            load_b_off(b0, pb, s * 16 * Kdim + k0);
            load_b_off(b1, pb, BCH + s * 16 * Kdim + k0);
            acc[s] = wmma(acc[s], ar, b0);
            acc[s] = wmma(acc[s], an, b1);
        }
    }
    const float lam = etas[it];                       // LAMBD == 1
    float part = 0.f, cnt = 0.f;
#pragma unroll
    for (int s = 0; s < 4; ++s) {
        const int c = colg * 64 + s * 16 + m;
#pragma unroll
        for (int j = 0; j < 8; ++j) {
            const int r = row0 + h8 + j;
            const float z  = acc[s][j];
            const float az = fabsf(z) - lam;
            const float sv = az > 0.f ? copysignf(az, z) : 0.f;
            s0bf[(size_t)r * Mdim + c] = f2bf(sv);
            const float asv = az > 0.f ? az : 0.f;
            part += asv;
            cnt  += (asv > 1e-3f) ? 1.f : 0.f;
        }
    }
    const float psum = blockSum(part, sh);
    if (threadIdx.x == 0) atomicAdd(loss_sparse, psum * (1.f / (float)Bdim));
    const float csum = blockSum(cnt, sh);
    if (threadIdx.x == 0 && last) atomicAdd(out_sparse, csum * (1.f / (float)Bdim));
}

// ---------------------------------------------------------------------------
// GEMM4: xmap_re = s0*E0T[0]^T, xmap_im = -(s0*E0T[1]^T); Huber(x_old, xmap);
//        x <- xmap (f32 + bf16).
// ---------------------------------------------------------------------------
__global__ void k_gemm4(const u16* __restrict__ s0bf, const u16* __restrict__ e0tbf,
                        float* __restrict__ x, u16* __restrict__ xbf,
                        float* __restrict__ loss_eq) {
    __shared__ float sh[256];
    const int tid = blockIdx.x * blockDim.x + threadIdx.x;
    const int wave = tid >> 5, lane = tid & 31;
    const int m = lane & 15, h8 = (lane >> 4) << 3, h16 = (lane >> 4) << 4;
    constexpr int NCOLG = Kdim / 64;
    const int colg = wave % NCOLG, row0 = (wave / NCOLG) * 16;

    constexpr int BCH = Kdim * Mdim;                  // E0T channel stride
    const u16* pa = s0bf + (size_t)(row0 + m) * Mdim + h8;
    const u16* pb = e0tbf + (size_t)(colg * 64 + m) * Mdim + h8;

    const v8f vz = {0.f,0.f,0.f,0.f,0.f,0.f,0.f,0.f};
    v8f accr[4], acci[4];
#pragma unroll
    for (int s = 0; s < 4; ++s) { accr[s] = vz; acci[s] = vz; }

#pragma unroll
    for (int k0 = 0; k0 < Mdim; k0 += 32) {
        Frag a;
        load_a_off(a, pa, k0);                        // one real A, reused 8x
#pragma unroll
        for (int s = 0; s < 4; ++s) {
            Frag b0, b1;
            load_b_off(b0, pb, s * 16 * Mdim + k0);
            load_b_off(b1, pb, BCH + s * 16 * Mdim + k0);
            accr[s] = wmma(accr[s], a, b0);
            acci[s] = wmma(acci[s], a, b1);
        }
    }
    float part = 0.f;
#pragma unroll
    for (int s = 0; s < 4; ++s) {
        const int c = colg * 64 + s * 16 + m;
#pragma unroll
        for (int j = 0; j < 8; ++j) {
            const int r = row0 + h8 + j;
            const size_t ir = (size_t)r * Kdim + c;
            const size_t ii = (size_t)Bdim * Kdim + ir;
            const float xmr = accr[s][j];
            const float xmi = -acci[s][j];
            const float dr = x[ir] - xmr, di = x[ii] - xmi;
            const float adr = fabsf(dr), adi = fabsf(di);
            part += (adr < 1.f ? 0.5f * dr * dr : adr - 0.5f);
            part += (adi < 1.f ? 0.5f * di * di : adi - 0.5f);
            x[ir] = xmr;  xbf[ir] = f2bf(xmr);
            x[ii] = xmi;  xbf[ii] = f2bf(xmi);
        }
    }
    const float psum = blockSum(part, sh);
    if (threadIdx.x == 0)
        atomicAdd(loss_eq, psum * (1.f / (2.f * (float)Kdim * (float)Bdim * (float)Titer)));
}

// ---------------------------------------------------------------------------
// Small utility kernels.
// ---------------------------------------------------------------------------
__global__ void k_cvt(const float* __restrict__ s, u16* __restrict__ d, int n) {
    int i = blockIdx.x * blockDim.x + threadIdx.x;
    if (i < n) d[i] = f2bf(s[i]);
}
__global__ void k_tre0(const float* __restrict__ e0, u16* __restrict__ e0t) {
    int i = blockIdx.x * blockDim.x + threadIdx.x;
    const int n = 2 * Kdim * Mdim;
    if (i < n) {
        const int ch = i / (Kdim * Mdim);
        const int rem = i % (Kdim * Mdim);
        const int k = rem / Mdim, mm = rem % Mdim;
        e0t[i] = f2bf(e0[(size_t)ch * Mdim * Kdim + (size_t)mm * Kdim + k]);
    }
}
__global__ void k_zero_x(float* __restrict__ x, u16* __restrict__ xbf, int n) {
    int i = blockIdx.x * blockDim.x + threadIdx.x;
    if (i < n) { x[i] = 0.f; xbf[i] = 0; }
}
__global__ void k_init3(float* __restrict__ p) {
    if (threadIdx.x < 3) p[threadIdx.x] = 0.f;
}
__global__ void k_copy(const float* __restrict__ s, float* __restrict__ d, int n) {
    int i = blockIdx.x * blockDim.x + threadIdx.x;
    if (i < n) d[i] = s[i];
}

// ---------------------------------------------------------------------------
extern "C" void kernel_launch(void* const* d_in, const int* in_sizes, int n_in,
                              void* d_out, int out_size, void* d_ws, size_t ws_size,
                              hipStream_t stream) {
    (void)in_sizes; (void)n_in; (void)out_size; (void)ws_size;

    const float* y      = (const float*)d_in[0];   // (B,2,N)
    const float* W      = (const float*)d_in[1];   // (2,N,K)
    const float* WX     = (const float*)d_in[2];   // (2,K,N)
    const float* E0     = (const float*)d_in[3];   // (2,M,K)
    const float* etas   = (const float*)d_in[4];   // (T+1)
    const float* gammas = (const float*)d_in[5];   // (T+1)

    float* out = (float*)d_out;                    // [2*B*K | ls | le | sp]

    // Workspace carve-out (256B aligned regions).
    char* ws = (char*)d_ws;
    size_t off = 0;
    auto take = [&](size_t bytes) -> char* {
        char* p = ws + off;
        off = (off + bytes + 255) & ~(size_t)255;
        return p;
    };
    u16*   wbf   = (u16*)  take((size_t)2 * Ndim * Kdim * 2);
    u16*   wxbf  = (u16*)  take((size_t)2 * Kdim * Ndim * 2);
    u16*   e0bf  = (u16*)  take((size_t)2 * Mdim * Kdim * 2);
    u16*   e0tbf = (u16*)  take((size_t)2 * Kdim * Mdim * 2);
    u16*   xbf   = (u16*)  take((size_t)2 * Bdim * Kdim * 2);
    u16*   tbf   = (u16*)  take((size_t)2 * Bdim * Ndim * 2);
    u16*   s0bf  = (u16*)  take((size_t)Bdim * Mdim * 2);
    float* xf    = (float*)take((size_t)2 * Bdim * Kdim * 4);

    const int BK = 2 * Bdim * Kdim;
    float* scal = out + BK;                        // 3 scalar outputs

    // One-time preprocessing.
    k_cvt   <<<(2 * Ndim * Kdim + 255) / 256, 256, 0, stream>>>(W,  wbf,  2 * Ndim * Kdim);
    k_cvt   <<<(2 * Kdim * Ndim + 255) / 256, 256, 0, stream>>>(WX, wxbf, 2 * Kdim * Ndim);
    k_cvt   <<<(2 * Mdim * Kdim + 255) / 256, 256, 0, stream>>>(E0, e0bf, 2 * Mdim * Kdim);
    k_tre0  <<<(2 * Kdim * Mdim + 255) / 256, 256, 0, stream>>>(E0, e0tbf);
    k_zero_x<<<(BK + 255) / 256, 256, 0, stream>>>(xf, xbf, BK);
    k_init3 <<<1, 64, 0, stream>>>(scal);

    // Wave-exact grids: 8 waves (256 threads) per block.
    const int g1 = (Bdim / 16) * (Ndim / 64) / 8;   // 128 blocks
    const int g2 = (Bdim / 16) * (Kdim / 64) / 8;   // 256 blocks
    const int g3 = (Bdim / 16) * (Mdim / 64) / 8;   // 1024 blocks
    const int g4 = (Bdim / 16) * (Kdim / 64) / 8;   // 256 blocks

    for (int it = 1; it <= Titer; ++it) {
        k_gemm1<<<g1, 256, 0, stream>>>(xbf, wbf, y, tbf);
        k_gemm2<<<g2, 256, 0, stream>>>(tbf, wxbf, gammas, it, xf, xbf);
        k_gemm3<<<g3, 256, 0, stream>>>(xbf, e0bf, etas, it, s0bf,
                                        scal + 0, scal + 2, (it == Titer) ? 1 : 0);
        k_gemm4<<<g4, 256, 0, stream>>>(s0bf, e0tbf, xf, xbf, scal + 1);
    }
    k_copy<<<(BK + 255) / 256, 256, 0, stream>>>(xf, out, BK);
}
